// MultiHeadSelfAttention_74105365725501
// MI455X (gfx1250) — compile-verified, matrix operand order checked by reading, not run on hardware
//
#include <hip/hip_runtime.h>

#define BATCH   8
#define CDIM    256
#define NPIX    1024
#define NHEADS  8
#define DHEAD   32
#define NPB     64    // pixels (columns) per projection block
#define XSTRIDE 264   // padded LDS row stride (halfs) to dodge bank conflicts

typedef float    v8f  __attribute__((ext_vector_type(8)));
typedef _Float16 v16h __attribute__((ext_vector_type(16)));

union ABFrag { unsigned int u[8]; v16h v; };
union H2     { unsigned int u; _Float16 h[2]; };

// K-pair base offset for VGPR v of an f16 A/B fragment (ISA 7.12.2):
// VGPR0..3 hold K = {0,1},{2,3},{4,5},{6,7} (+8 for lanes 16-31),
// VGPR4..7 hold K = {16,17}..{22,23} (+8 for lanes 16-31).
__device__ __forceinline__ int kb_of(int v) { return 2 * v + ((v >= 4) ? 8 : 0); }

__device__ __forceinline__ v8f v8f_zero() {
    v8f z;
#pragma unroll
    for (int i = 0; i < 8; i++) z[i] = 0.0f;
    return z;
}

// ---------------------------------------------------------------------------
// Kernel 1: fused QKV projection.  Out = W * X + b  (per batch, X is C x N).
// grid = (N/NPB, BATCH, 3proj), block = 256 (8 waves).
// Wave w computes rows [32w, 32w+32) over the 64-pixel tile (2x4 WMMA tiles).
// Q,K written as [b][h][n][d] f16 (Q pre-scaled by 1/sqrt(d)); V as [b][h][d][n].
// ---------------------------------------------------------------------------
__global__ __launch_bounds__(256) void mhsa_qkv_proj(
    const float* __restrict__ x,
    const float* __restrict__ Wq, const float* __restrict__ bq,
    const float* __restrict__ Wk, const float* __restrict__ bk,
    const float* __restrict__ Wv, const float* __restrict__ bv,
    _Float16* __restrict__ q_ws, _Float16* __restrict__ k_ws,
    _Float16* __restrict__ v_ws)
{
    __shared__ _Float16 xt[NPB * XSTRIDE];  // X tile transposed: [n_local][c] f16

    const int nbase = blockIdx.x * NPB;
    const int b     = blockIdx.y;
    const int proj  = blockIdx.z;
    const float* W    = (proj == 0) ? Wq : ((proj == 1) ? Wk : Wv);
    const float* bias = (proj == 0) ? bq : ((proj == 1) ? bk : bv);
    const float* xb   = x + (size_t)b * CDIM * NPIX;
    const int tid = threadIdx.x;

    for (int i = tid; i < CDIM * NPB; i += 256) {
        const int c  = i >> 6;
        const int nl = i & (NPB - 1);
        xt[nl * XSTRIDE + c] = (_Float16)xb[c * NPIX + nbase + nl];
    }
    __syncthreads();

    const int wave = tid >> 5, lane = tid & 31;
    const int hi = lane >> 4, lr = lane & 15;
    const int Mt0 = wave * 32;

    v8f acc[2][4];
#pragma unroll
    for (int t = 0; t < 2; t++)
#pragma unroll
        for (int s = 0; s < 4; s++) acc[t][s] = v8f_zero();

    for (int K0 = 0; K0 < CDIM; K0 += 32) {
        ABFrag af[2];
#pragma unroll
        for (int t = 0; t < 2; t++) {
            const float* wr = W + (Mt0 + t * 16 + lr) * CDIM + K0 + hi * 8;
#pragma unroll
            for (int v = 0; v < 8; v++) {
                const float2 wp = *(const float2*)(wr + kb_of(v));
                H2 cv; cv.h[0] = (_Float16)wp.x; cv.h[1] = (_Float16)wp.y;
                af[t].u[v] = cv.u;
            }
        }
        ABFrag bf[4];
#pragma unroll
        for (int s = 0; s < 4; s++) {
            const _Float16* bp = &xt[(s * 16 + lr) * XSTRIDE + K0 + hi * 8];
#pragma unroll
            for (int v = 0; v < 8; v++)
                bf[s].u[v] = *(const unsigned int*)(bp + kb_of(v));
        }
#pragma unroll
        for (int t = 0; t < 2; t++)
#pragma unroll
            for (int s = 0; s < 4; s++)
                acc[t][s] = __builtin_amdgcn_wmma_f32_16x16x32_f16(
                    false, af[t].v, false, bf[s].v, (short)0, acc[t][s], false, false);
    }

    const float scale = (proj == 0) ? 0.17677669529663687f : 1.0f;  // 1/sqrt(32)
#pragma unroll
    for (int t = 0; t < 2; t++) {
#pragma unroll
        for (int s = 0; s < 4; s++) {
            const int n = nbase + s * 16 + lr;
#pragma unroll
            for (int r = 0; r < 8; r++) {
                const int row = Mt0 + t * 16 + r + hi * 8;
                const _Float16 hval = (_Float16)((acc[t][s][r] + bias[row]) * scale);
                const int head = row >> 5, dd = row & 31;
                if (proj == 2)
                    v_ws[((size_t)(b * NHEADS + head) * DHEAD + dd) * NPIX + n] = hval;
                else if (proj == 0)
                    q_ws[((size_t)(b * NHEADS + head) * NPIX + n) * DHEAD + dd] = hval;
                else
                    k_ws[((size_t)(b * NHEADS + head) * NPIX + n) * DHEAD + dd] = hval;
            }
        }
    }
}

// ---------------------------------------------------------------------------
// Kernel 2: flash attention.  grid = (N/128, B*H), block = 256 (8 waves).
// Each wave owns 16 query rows; iterates keys in chunks of 32:
//   2x score WMMA (K = d = 32) -> register softmax (shfl reductions over the
//   16-lane column groups) -> P relayout via wave-private LDS -> 2x PV WMMA.
// Output stored [b][n][c] f16.
// ---------------------------------------------------------------------------
__global__ __launch_bounds__(256) void mhsa_flash_attn(
    const _Float16* __restrict__ q_ws, const _Float16* __restrict__ k_ws,
    const _Float16* __restrict__ v_ws, _Float16* __restrict__ attn_ws)
{
    __shared__ _Float16 pbuf[8][16][40];  // per-wave 16x32 P tile, padded

    const int bh = blockIdx.y;
    const int b = bh >> 3, head = bh & 7;
    const int tid = threadIdx.x, wave = tid >> 5, lane = tid & 31;
    const int hi = lane >> 4, lr = lane & 15;
    const int qbase = blockIdx.x * 128 + wave * 16;

    ABFrag qf;
    {
        const _Float16* qp = q_ws + ((size_t)bh * NPIX + qbase + lr) * DHEAD + hi * 8;
#pragma unroll
        for (int v = 0; v < 8; v++) qf.u[v] = *(const unsigned int*)(qp + kb_of(v));
    }

    v8f acc0 = v8f_zero(), acc1 = v8f_zero();
    float mrow[8], lrow[8];
#pragma unroll
    for (int r = 0; r < 8; r++) { mrow[r] = -1e30f; lrow[r] = 0.0f; }

    for (int m0 = 0; m0 < NPIX; m0 += 32) {
        v8f s[2];
#pragma unroll
        for (int t = 0; t < 2; t++) {
            ABFrag kf;
            const _Float16* kp = k_ws + ((size_t)bh * NPIX + m0 + t * 16 + lr) * DHEAD + hi * 8;
#pragma unroll
            for (int v = 0; v < 8; v++) kf.u[v] = *(const unsigned int*)(kp + kb_of(v));
            s[t] = __builtin_amdgcn_wmma_f32_16x16x32_f16(
                false, qf.v, false, kf.v, (short)0, v8f_zero(), false, false);
        }

#pragma unroll
        for (int r = 0; r < 8; r++) {
            float mx = fmaxf(s[0][r], s[1][r]);
            mx = fmaxf(mx, __shfl_xor(mx, 1, 32));
            mx = fmaxf(mx, __shfl_xor(mx, 2, 32));
            mx = fmaxf(mx, __shfl_xor(mx, 4, 32));
            mx = fmaxf(mx, __shfl_xor(mx, 8, 32));
            const float mnew  = fmaxf(mrow[r], mx);
            const float alpha = exp2f((mrow[r] - mnew) * 1.4426950408889634f);
            mrow[r] = mnew;
            const float p0 = exp2f((s[0][r] - mnew) * 1.4426950408889634f);
            const float p1 = exp2f((s[1][r] - mnew) * 1.4426950408889634f);
            float rs = p0 + p1;
            rs += __shfl_xor(rs, 1, 32);
            rs += __shfl_xor(rs, 2, 32);
            rs += __shfl_xor(rs, 4, 32);
            rs += __shfl_xor(rs, 8, 32);
            lrow[r] = lrow[r] * alpha + rs;
            acc0[r] *= alpha;
            acc1[r] *= alpha;
            pbuf[wave][r + hi * 8][lr]      = (_Float16)p0;
            pbuf[wave][r + hi * 8][16 + lr] = (_Float16)p1;
        }
        asm volatile("s_wait_dscnt 0x0" ::: "memory");  // wave-private LDS RAW

        ABFrag pf;
        const _Float16* pp = &pbuf[wave][lr][hi * 8];
#pragma unroll
        for (int v = 0; v < 8; v++) pf.u[v] = *(const unsigned int*)(pp + kb_of(v));

#pragma unroll
        for (int t = 0; t < 2; t++) {
            ABFrag vf;
            const _Float16* vp = v_ws + ((size_t)bh * DHEAD + t * 16 + lr) * NPIX + m0 + hi * 8;
#pragma unroll
            for (int v = 0; v < 8; v++) vf.u[v] = *(const unsigned int*)(vp + kb_of(v));
            if (t == 0)
                acc0 = __builtin_amdgcn_wmma_f32_16x16x32_f16(
                    false, pf.v, false, vf.v, (short)0, acc0, false, false);
            else
                acc1 = __builtin_amdgcn_wmma_f32_16x16x32_f16(
                    false, pf.v, false, vf.v, (short)0, acc1, false, false);
        }
    }

#pragma unroll
    for (int r = 0; r < 8; r++) {
        const int n = qbase + r + hi * 8;
        const float inv = 1.0f / lrow[r];
        _Float16* dst = attn_ws + ((size_t)b * NPIX + n) * CDIM + head * DHEAD;
        dst[lr]      = (_Float16)(acc0[r] * inv);
        dst[16 + lr] = (_Float16)(acc1[r] * inv);
    }
}

// ---------------------------------------------------------------------------
// Kernel 3: output projection + bias + residual.  grid = (N/NPB, BATCH).
// B fragments read directly from attn_ws [b][n][c] (K-pairs contiguous).
// ---------------------------------------------------------------------------
__global__ __launch_bounds__(256) void mhsa_out_proj(
    const _Float16* __restrict__ attn_ws, const float* __restrict__ Wo,
    const float* __restrict__ bo, const float* __restrict__ x,
    float* __restrict__ out)
{
    const int nbase = blockIdx.x * NPB;
    const int b = blockIdx.y;
    const int tid = threadIdx.x, wave = tid >> 5, lane = tid & 31;
    const int hi = lane >> 4, lr = lane & 15;
    const int Mt0 = wave * 32;

    v8f acc[2][4];
#pragma unroll
    for (int t = 0; t < 2; t++)
#pragma unroll
        for (int s = 0; s < 4; s++) acc[t][s] = v8f_zero();

    for (int K0 = 0; K0 < CDIM; K0 += 32) {
        ABFrag af[2];
#pragma unroll
        for (int t = 0; t < 2; t++) {
            const float* wr = Wo + (Mt0 + t * 16 + lr) * CDIM + K0 + hi * 8;
#pragma unroll
            for (int v = 0; v < 8; v++) {
                const float2 wp = *(const float2*)(wr + kb_of(v));
                H2 cv; cv.h[0] = (_Float16)wp.x; cv.h[1] = (_Float16)wp.y;
                af[t].u[v] = cv.u;
            }
        }
        ABFrag bf[4];
#pragma unroll
        for (int s = 0; s < 4; s++) {
            const _Float16* bp = attn_ws + ((size_t)b * NPIX + nbase + s * 16 + lr) * CDIM + K0 + hi * 8;
#pragma unroll
            for (int v = 0; v < 8; v++)
                bf[s].u[v] = *(const unsigned int*)(bp + kb_of(v));
        }
#pragma unroll
        for (int t = 0; t < 2; t++)
#pragma unroll
            for (int s = 0; s < 4; s++)
                acc[t][s] = __builtin_amdgcn_wmma_f32_16x16x32_f16(
                    false, af[t].v, false, bf[s].v, (short)0, acc[t][s], false, false);
    }

#pragma unroll
    for (int t = 0; t < 2; t++) {
#pragma unroll
        for (int s = 0; s < 4; s++) {
            const int n = nbase + s * 16 + lr;
#pragma unroll
            for (int r = 0; r < 8; r++) {
                const int row = Mt0 + t * 16 + r + hi * 8;
                const size_t idx = ((size_t)b * CDIM + row) * NPIX + n;
                out[idx] = acc[t][s][r] + bo[row] + x[idx];
            }
        }
    }
}

extern "C" void kernel_launch(void* const* d_in, const int* in_sizes, int n_in,
                              void* d_out, int out_size, void* d_ws, size_t ws_size,
                              hipStream_t stream) {
    (void)in_sizes; (void)n_in; (void)out_size; (void)ws_size;
    const float* x  = (const float*)d_in[0];
    const float* Wq = (const float*)d_in[1];
    const float* bq = (const float*)d_in[2];
    const float* Wk = (const float*)d_in[3];
    const float* bk = (const float*)d_in[4];
    const float* Wv = (const float*)d_in[5];
    const float* bv = (const float*)d_in[6];
    const float* Wo = (const float*)d_in[7];
    const float* bo = (const float*)d_in[8];
    float* out = (float*)d_out;

    const size_t qkv_elems = (size_t)BATCH * NHEADS * NPIX * DHEAD;  // 2M halfs
    _Float16* q_ws    = (_Float16*)d_ws;
    _Float16* k_ws    = q_ws + qkv_elems;
    _Float16* v_ws    = k_ws + qkv_elems;
    _Float16* attn_ws = v_ws + qkv_elems;  // [b][n][c], 2M halfs

    mhsa_qkv_proj<<<dim3(NPIX / NPB, BATCH, 3), 256, 0, stream>>>(
        x, Wq, bq, Wk, bk, Wv, bv, q_ws, k_ws, v_ws);
    mhsa_flash_attn<<<dim3(NPIX / 128, BATCH * NHEADS), 256, 0, stream>>>(
        q_ws, k_ws, v_ws, attn_ws);
    mhsa_out_proj<<<dim3(NPIX / NPB, BATCH), 256, 0, stream>>>(
        attn_ws, Wo, bo, x, out);
}
